// GeneralConvNet_51754355916840
// MI455X (gfx1250) — matrix operand order repaired and checked by reading, not compiled, not run on hardware
//
#include <hip/hip_runtime.h>
#include <cstdint>
#include <cstddef>

#define HEADS 4
#define NEG_INF_KEY 0x007FFFFFu  // fkey(-inf)

typedef __attribute__((ext_vector_type(16))) _Float16 v16h;
typedef __attribute__((ext_vector_type(8)))  float    v8f;

union HV { v16h v; unsigned int u[8]; };

__device__ __forceinline__ float lrelu(float x) { return x > 0.f ? x : 0.2f * x; }

// order-preserving float -> uint key (monotone for all finite floats)
__device__ __forceinline__ unsigned int fkey(float f) {
  unsigned int u = __float_as_uint(f);
  return (u & 0x80000000u) ? ~u : (u | 0x80000000u);
}
__device__ __forceinline__ float fkey_dec(unsigned int k) {
  return (k & 0x80000000u) ? __uint_as_float(k ^ 0x80000000u) : __uint_as_float(~k);
}

// ---------------------------------------------------------------------------
// WMMA GEMM: C[M x NC] = A[M x K](f16,row-major) @ Bt[NC x K](f16, K-contig
// per output column, i.e. B transposed) + bias[NC].  One wave per 16x16 tile.
// K % 32 == 0, NC % 16 == 0.  Rows clamped for loads, predicated for stores,
// so EXEC stays all-ones through the WMMA (ISA 7.12 requirement).
// ---------------------------------------------------------------------------
__global__ void __launch_bounds__(32)
wmma_gemm(const _Float16* __restrict__ A, const _Float16* __restrict__ Bt,
          const float* __restrict__ bias, float* __restrict__ C,
          int M, int K, int NC) {
  const int lane = threadIdx.x;   // 0..31 (wave32)
  const int hi   = lane >> 4;     // 0 = K-low lane half, 1 = K-high lane half
  const int l16  = lane & 15;
  const int tn = blockIdx.x, tm = blockIdx.y;

  int arow = tm * 16 + l16; if (arow >= M) arow = M - 1;   // clamp (M%16==0 normally)
  const int bcol = tn * 16 + l16;

  v8f acc = {};
  for (int k0 = 0; k0 < K; k0 += 32) {
    HV a, b;
    const unsigned int* ap = reinterpret_cast<const unsigned int*>(A + (size_t)arow * K + k0);
    const unsigned int* bp = reinterpret_cast<const unsigned int*>(Bt + (size_t)bcol * K + k0);
    // A 16x32 f16 layout: lanes<16 hold K pairs {0,2,4,6} then {16,18,20,22};
    // lanes>=16 shifted by +8  (ISA 7.12.2, 16-bit A-matrix table)
#pragma unroll
    for (int v = 0; v < 4; ++v) a.u[v]     = ap[v + hi * 4];
#pragma unroll
    for (int v = 0; v < 4; ++v) a.u[4 + v] = ap[8 + v + hi * 4];
    // B 32x16 f16 layout: col = lane&15; lanes<16 K=0..7,16..23, lanes>=16 K=8..15,24..31
#pragma unroll
    for (int v = 0; v < 4; ++v) b.u[v]     = bp[v + hi * 4];
#pragma unroll
    for (int v = 0; v < 4; ++v) b.u[4 + v] = bp[8 + v + hi * 4];

    acc = __builtin_amdgcn_wmma_f32_16x16x32_f16(
        /*neg_a=*/false, a.v, /*neg_b=*/false, b.v,
        /*c_mod=*/(short)0, acc, /*reuse_a=*/false, /*reuse_b=*/false);
  }
  const float bv = bias ? bias[bcol] : 0.f;
  // C/D layout: VGPR r -> row M = r + 8*hi, col N = lane&15
#pragma unroll
  for (int r = 0; r < 8; ++r) {
    int m = tm * 16 + hi * 8 + r;
    if (m < M) C[(size_t)m * NC + bcol] = acc[r] + bv;
  }
}

// ---------------------------------------------------------------------------
// Support kernels
// ---------------------------------------------------------------------------
__global__ void fill_f32(float* p, float v, size_t n) {
  size_t i = (size_t)blockIdx.x * blockDim.x + threadIdx.x;
  if (i < n) p[i] = v;
}
__global__ void fill_f32_v4(float4* p, float v, size_t n4) {
  size_t i = (size_t)blockIdx.x * blockDim.x + threadIdx.x;
  if (i < n4) p[i] = make_float4(v, v, v, v);
}
__global__ void fill_u32(unsigned int* p, unsigned int v, size_t n) {
  size_t i = (size_t)blockIdx.x * blockDim.x + threadIdx.x;
  if (i < n) p[i] = v;
}
// f32 -> f16, 4 elements per thread (b128 in, b64 out)
__global__ void f32_to_f16_v4(const float* __restrict__ in, _Float16* __restrict__ out, size_t n4) {
  size_t i = (size_t)blockIdx.x * blockDim.x + threadIdx.x;
  if (i >= n4) return;
  float4 vv = ((const float4*)in)[i];
  _Float16 h4[4] = { (_Float16)vv.x, (_Float16)vv.y, (_Float16)vv.z, (_Float16)vv.w };
  *(uint2*)(out + i * 4) = *(uint2*)h4;
}
// W[K x NC] f32 -> Wt[NC x K] f16 (K contiguous per output column)
__global__ void wT_f16(const float* __restrict__ W, _Float16* __restrict__ Wt, int K, int NC) {
  int i = blockIdx.x * blockDim.x + threadIdx.x;
  if (i >= K * NC) return;
  int n = i / K, k = i % K;
  Wt[(size_t)n * K + k] = (_Float16)W[(size_t)k * NC + n];
}

// per-node attention logits: s[n,h] = sum_c msg[n, h*C+c] * att[h,c]
__global__ void att_score(const float* __restrict__ msg, const float* __restrict__ att,
                          float* __restrict__ s, int N, int C) {
  int i = blockIdx.x * blockDim.x + threadIdx.x;
  if (i >= N * HEADS) return;
  int n = i >> 2, h = i & 3;
  const float* mrow = msg + (size_t)n * HEADS * C + (size_t)h * C;
  const float* arow = att + (size_t)h * C;
  float acc = 0.f;
  for (int c = 0; c < C; ++c) acc += mrow[c] * arow[c];
  s[i] = acc;
}

__global__ void edge_amax(const int* __restrict__ src, const int* __restrict__ dst,
                          const float* __restrict__ s, unsigned int* __restrict__ amaxk, int E) {
  int i = blockIdx.x * blockDim.x + threadIdx.x;
  if (i >= E * HEADS) return;
  int e = i >> 2, h = i & 3;
  float a = lrelu(s[(size_t)src[e] * HEADS + h]);
  atomicMax(&amaxk[(size_t)dst[e] * HEADS + h], fkey(a));
}

__global__ void amax_decode(const unsigned int* __restrict__ k, float* __restrict__ amax, size_t n) {
  size_t i = (size_t)blockIdx.x * blockDim.x + threadIdx.x;
  if (i >= n) return;
  unsigned int kk = k[i];
  amax[i] = (kk == NEG_INF_KEY) ? 0.f : fkey_dec(kk);   // empty segment -> 0 (reference)
}

__global__ void edge_expsum(const int* __restrict__ src, const int* __restrict__ dst,
                            const float* __restrict__ s, const float* __restrict__ amax,
                            float* __restrict__ asum, int E) {
  int i = blockIdx.x * blockDim.x + threadIdx.x;
  if (i >= E * HEADS) return;
  int e = i >> 2, h = i & 3;
  float a = lrelu(s[(size_t)src[e] * HEADS + h]);
  atomicAdd(&asum[(size_t)dst[e] * HEADS + h], __expf(a - amax[(size_t)dst[e] * HEADS + h]));
}

// heavy pass: agg[dst, h*C+c] += msg[src, h*C+c] * softmax_w(e,h).
// One block per edge, blockDim = HEADS*C/4; each thread moves a float4
// (global_load_b128) and issues 4 atomic adds.  The 4 per-head weights are
// computed once by lanes 0..3 and broadcast through LDS.  msg/agg (76.8MB
// each) are L2-resident (192MB L2), so the atomics stay on-chip.
__global__ void __launch_bounds__(128)
edge_scatter(const int* __restrict__ src, const int* __restrict__ dst,
             const float* __restrict__ s, const float* __restrict__ amax,
             const float* __restrict__ asum, const float* __restrict__ msg,
             float* __restrict__ agg, int C) {
  __shared__ float wsm[HEADS];
  const int e = blockIdx.x;
  const int sn = src[e], dn = dst[e];     // uniform -> scalar loads
  const int t = threadIdx.x;
  if (t < HEADS) {
    float a = lrelu(s[(size_t)sn * HEADS + t]);
    wsm[t] = __expf(a - amax[(size_t)dn * HEADS + t]) / (asum[(size_t)dn * HEADS + t] + 1e-16f);
  }
  __syncthreads();
  const int c4 = t * 4;                   // 4 consecutive channels, same head (C%4==0)
  const float w = wsm[c4 / C];
  const float4 mv = *(const float4*)(msg + (size_t)sn * HEADS * C + c4);
  float* ag = agg + (size_t)dn * HEADS * C + c4;
  atomicAdd(ag + 0, mv.x * w);
  atomicAdd(ag + 1, mv.y * w);
  atomicAdd(ag + 2, mv.z * w);
  atomicAdd(ag + 3, mv.w * w);
}

// out[n,c] = mean_h agg[n,h,c] + selfv[n,c], float4 per thread
__global__ void head_mean_add(const float* __restrict__ agg, const float* __restrict__ selfv,
                              float* __restrict__ out, int N, int C) {
  const int Cq = C >> 2;
  size_t i = (size_t)blockIdx.x * blockDim.x + threadIdx.x;
  if (i >= (size_t)N * Cq) return;
  size_t n = i / Cq; int c4 = (int)(i % Cq) << 2;
  const float* ar = agg + n * (size_t)(HEADS * C);
  float4 a0 = *(const float4*)(ar + c4);
  float4 a1 = *(const float4*)(ar + C + c4);
  float4 a2 = *(const float4*)(ar + 2 * C + c4);
  float4 a3 = *(const float4*)(ar + 3 * C + c4);
  float4 sv = *(const float4*)(selfv + n * (size_t)C + c4);
  float4 o;
  o.x = 0.25f * (a0.x + a1.x + a2.x + a3.x) + sv.x;
  o.y = 0.25f * (a0.y + a1.y + a2.y + a3.y) + sv.y;
  o.z = 0.25f * (a0.z + a1.z + a2.z + a3.z) + sv.z;
  o.w = 0.25f * (a0.w + a1.w + a2.w + a3.w) + sv.w;
  *(float4*)(out + n * (size_t)C + c4) = o;
}

__global__ void pool_scatter(const float* __restrict__ h, const int* __restrict__ batch,
                             float* __restrict__ pooled, float* __restrict__ cnt, int N, int C) {
  const int Cq = C >> 2;
  size_t i = (size_t)blockIdx.x * blockDim.x + threadIdx.x;
  if (i >= (size_t)N * Cq) return;
  size_t n = i / Cq; int c4 = (int)(i % Cq) << 2;
  int g = batch[n];
  float4 v = *(const float4*)(h + n * (size_t)C + c4);
  float* pg = pooled + (size_t)g * C + c4;
  atomicAdd(pg + 0, v.x);
  atomicAdd(pg + 1, v.y);
  atomicAdd(pg + 2, v.z);
  atomicAdd(pg + 3, v.w);
  if (c4 == 0) atomicAdd(&cnt[g], 1.f);
}

__global__ void classify(const float* __restrict__ pooled, const float* __restrict__ cnt,
                         const float* __restrict__ Wfc, const float* __restrict__ bfc,
                         float* __restrict__ out, int G, int C, int OUT) {
  int g = blockIdx.x * blockDim.x + threadIdx.x;
  if (g >= G || OUT > 16) return;
  float inv = 1.0f / fmaxf(cnt[g], 1.0f);
  float l[16];
  float m = -3.0e38f;
  for (int j = 0; j < OUT; ++j) {
    float acc = bfc[j];
    for (int k = 0; k < C; ++k) acc += (pooled[(size_t)g * C + k] * inv) * Wfc[(size_t)k * OUT + j];
    l[j] = acc; m = fmaxf(m, acc);
  }
  float ssum = 0.f;
  for (int j = 0; j < OUT; ++j) ssum += __expf(l[j] - m);
  float lse = m + logf(ssum);
  for (int j = 0; j < OUT; ++j) out[(size_t)g * OUT + j] = l[j] - lse;
}

// ---------------------------------------------------------------------------
static inline unsigned int cdivu(size_t a, size_t b) { return (unsigned int)((a + b - 1) / b); }

extern "C" void kernel_launch(void* const* d_in, const int* in_sizes, int n_in,
                              void* d_out, int out_size, void* d_ws, size_t ws_size,
                              hipStream_t stream) {
  (void)n_in; (void)ws_size;
  const int F_IN = 64, H1 = 96, C3 = 64, OUT = 10;
  const int N = in_sizes[0] / F_IN;
  const int E = in_sizes[1] / 2;
  const int G = out_size / OUT;

  const float* x    = (const float*)d_in[0];
  const int*   ei   = (const int*)d_in[1];
  const int*   batch= (const int*)d_in[2];
  const float* Wm1  = (const float*)d_in[3];
  const float* bm1  = (const float*)d_in[4];
  const float* att1 = (const float*)d_in[5];
  const float* Ws1  = (const float*)d_in[6];
  const float* bs1  = (const float*)d_in[7];
  const float* Wm2  = (const float*)d_in[8];
  const float* bm2  = (const float*)d_in[9];
  const float* att2 = (const float*)d_in[10];
  const float* Wm3  = (const float*)d_in[11];
  const float* bm3  = (const float*)d_in[12];
  const float* att3 = (const float*)d_in[13];
  const float* Ws3  = (const float*)d_in[14];
  const float* bs3  = (const float*)d_in[15];
  const float* Wfc  = (const float*)d_in[16];
  const float* bfc  = (const float*)d_in[17];
  const int* srcI = ei;
  const int* dstI = ei + E;

  // workspace carve (256B aligned)
  char* p = (char*)d_ws;
  auto carve = [&](size_t bytes) -> void* {
    void* r = (void*)p; p += (bytes + 255) & ~(size_t)255; return r;
  };
  _Float16* feat16 = (_Float16*)carve((size_t)N * 96 * sizeof(_Float16));
  _Float16* wTm    = (_Float16*)carve((size_t)96 * 384 * sizeof(_Float16));
  _Float16* wTs    = (_Float16*)carve((size_t)96 * 96 * sizeof(_Float16));
  float* msg   = (float*)carve((size_t)N * 384 * sizeof(float));
  float* agg   = (float*)carve((size_t)N * 384 * sizeof(float));
  float* sbuf  = (float*)carve((size_t)N * HEADS * sizeof(float));
  unsigned int* amk = (unsigned int*)carve((size_t)N * HEADS * sizeof(unsigned int));
  float* amax  = (float*)carve((size_t)N * HEADS * sizeof(float));
  float* asum  = (float*)carve((size_t)N * HEADS * sizeof(float));
  float* selfb = (float*)carve((size_t)N * 96 * sizeof(float));
  float* hA    = (float*)carve((size_t)N * 96 * sizeof(float));
  float* hB    = (float*)carve((size_t)N * 96 * sizeof(float));
  float* pooled= (float*)carve((size_t)G * C3 * sizeof(float));
  float* cnt   = (float*)carve((size_t)G * sizeof(float));

  auto run_layer = [&](const float* featf32, int K,
                       const float* Wm, const float* bm, const float* att, int C,
                       const float* Ws, const float* bs, float* outBuf) {
    const int HC = HEADS * C;
    size_t nK4 = (size_t)N * K / 4;
    f32_to_f16_v4<<<cdivu(nK4, 256), 256, 0, stream>>>(featf32, feat16, nK4);
    wT_f16<<<cdivu((size_t)K * HC, 256), 256, 0, stream>>>(Wm, wTm, K, HC);
    wmma_gemm<<<dim3(HC / 16, (N + 15) / 16), 32, 0, stream>>>(feat16, wTm, bm, msg, N, K, HC);

    const float* selfsrc;
    if (Ws) {
      wT_f16<<<cdivu((size_t)K * C, 256), 256, 0, stream>>>(Ws, wTs, K, C);
      wmma_gemm<<<dim3(C / 16, (N + 15) / 16), 32, 0, stream>>>(feat16, wTs, bs, selfb, N, K, C);
      selfsrc = selfb;
    } else {
      selfsrc = featf32;  // lin_self = Identity (in==out dims)
    }

    att_score<<<cdivu((size_t)N * HEADS, 256), 256, 0, stream>>>(msg, att, sbuf, N, C);
    fill_u32<<<cdivu((size_t)N * HEADS, 256), 256, 0, stream>>>(amk, NEG_INF_KEY, (size_t)N * HEADS);
    fill_f32<<<cdivu((size_t)N * HEADS, 256), 256, 0, stream>>>(asum, 0.f, (size_t)N * HEADS);
    fill_f32_v4<<<cdivu((size_t)N * HC / 4, 256), 256, 0, stream>>>((float4*)agg, 0.f, (size_t)N * HC / 4);

    edge_amax<<<cdivu((size_t)E * HEADS, 256), 256, 0, stream>>>(srcI, dstI, sbuf, amk, E);
    amax_decode<<<cdivu((size_t)N * HEADS, 256), 256, 0, stream>>>(amk, amax, (size_t)N * HEADS);
    edge_expsum<<<cdivu((size_t)E * HEADS, 256), 256, 0, stream>>>(srcI, dstI, sbuf, amax, asum, E);
    edge_scatter<<<E, HC / 4, 0, stream>>>(srcI, dstI, sbuf, amax, asum, msg, agg, C);
    head_mean_add<<<cdivu((size_t)N * C / 4, 256), 256, 0, stream>>>(agg, selfsrc, outBuf, N, C);
  };

  run_layer(x,  F_IN, Wm1, bm1, att1, H1, Ws1,     bs1,     hA);
  run_layer(hA, H1,   Wm2, bm2, att2, H1, nullptr, nullptr, hB);
  run_layer(hB, H1,   Wm3, bm3, att3, C3, Ws3,     bs3,     hA);

  fill_f32<<<cdivu((size_t)G * C3, 256), 256, 0, stream>>>(pooled, 0.f, (size_t)G * C3);
  fill_f32<<<cdivu((size_t)G, 64), 64, 0, stream>>>(cnt, 0.f, (size_t)G);
  pool_scatter<<<cdivu((size_t)N * C3 / 4, 256), 256, 0, stream>>>(hA, batch, pooled, cnt, N, C3);
  classify<<<cdivu((size_t)G, 64), 64, 0, stream>>>(pooled, cnt, Wfc, bfc, (float*)d_out, G, C3, OUT);
}
